// OmegaBlock_26130581029533
// MI455X (gfx1250) — 16-bit and 8-bit matrix operand order repaired and checked by reading, compiled, not run on hardware
//
#include <hip/hip_runtime.h>
#include <hip/hip_bf16.h>
#include <cstdint>


typedef __attribute__((ext_vector_type(8)))  int      v8i_t;
typedef __attribute__((ext_vector_type(8)))  float    v8f_t;
typedef __attribute__((ext_vector_type(16))) _Float16 v16h_t;

static constexpr int Bc  = 2;
static constexpr int Sc  = 2048;
static constexpr int Hc  = 1024;
static constexpr int NHc = 16;
static constexpr int Ec  = 8;
static constexpr int Dc  = 1024;
static constexpr int HDc = 64;
static constexpr int Tc  = Bc * Sc;
static constexpr int PB  = 256;
static constexpr int KD  = 1024;

union FragI8 { v8i_t v; int i[8]; };
union AccI   { v8i_t v; int i[8]; };
union AccF   { v8f_t v; float f[8]; };
union FragH  { v16h_t v; _Float16 h[16]; unsigned u[8]; uint4 q[2]; };

__global__ void __launch_bounds__(256)
absmean_partial(const float* __restrict__ w, long elemsPerTensor,
                float* __restrict__ partials) {
  const int t = blockIdx.y, tid = threadIdx.x;
  const float* src = w + (size_t)t * elemsPerTensor;
  const long chunk = elemsPerTensor / PB;
  const long base  = (long)blockIdx.x * chunk;
  float s = 0.f;
  for (long i = base + tid; i < base + chunk; i += 256) s += fabsf(src[i]);
  __shared__ float red[256];
  red[tid] = s; __syncthreads();
  for (int o = 128; o > 0; o >>= 1) { if (tid < o) red[tid] += red[tid + o]; __syncthreads(); }
  if (tid == 0) partials[t * PB + blockIdx.x] = red[0];
}

__global__ void __launch_bounds__(256)
make_scales(const float* __restrict__ partials, float invCount,
            float* __restrict__ scales) {
  const int t = blockIdx.x, tid = threadIdx.x;
  __shared__ float red[256];
  red[tid] = partials[t * PB + tid]; __syncthreads();
  for (int o = 128; o > 0; o >>= 1) { if (tid < o) red[tid] += red[tid + o]; __syncthreads(); }
  if (tid == 0) scales[t] = 1.f / fmaxf(red[0] * invCount, 1e-5f);
}

__global__ void __launch_bounds__(256)
quantize_w(const float* __restrict__ w, const float* __restrict__ scales,
           long elemsPerTensor, long total, int8_t* __restrict__ out) {
  const long stride = (long)gridDim.x * blockDim.x;
  for (long i = (long)blockIdx.x * blockDim.x + threadIdx.x; i < total; i += stride) {
    float s = scales[i / elemsPerTensor];
    float q = rintf(w[i] * s);
    q = fminf(1.f, fmaxf(-1.f, q));
    out[i] = (int8_t)q;
  }
}

__global__ void __launch_bounds__(256)
rmsnorm_quant(const float* __restrict__ x, const float* __restrict__ g,
              float* __restrict__ xnOut, int writeXn,
              int8_t* __restrict__ xq, float* __restrict__ sA) {
  const int row = blockIdx.x, tid = threadIdx.x;
  const float* xr = x + (size_t)row * Hc;
  float v[4]; float ss = 0.f;
#pragma unroll
  for (int j = 0; j < 4; ++j) { v[j] = xr[tid + 256 * j]; ss += v[j] * v[j]; }
  __shared__ float red[256];
  red[tid] = ss; __syncthreads();
  for (int o = 128; o > 0; o >>= 1) { if (tid < o) red[tid] += red[tid + o]; __syncthreads(); }
  const float r = rsqrtf(red[0] * (1.f / Hc) + 1e-5f);
  __syncthreads();
  float xn[4]; float am = 0.f;
#pragma unroll
  for (int j = 0; j < 4; ++j) { xn[j] = v[j] * r * g[tid + 256 * j]; am = fmaxf(am, fabsf(xn[j])); }
  red[tid] = am; __syncthreads();
  for (int o = 128; o > 0; o >>= 1) { if (tid < o) red[tid] = fmaxf(red[tid], red[tid + o]); __syncthreads(); }
  const float s = 127.f / fmaxf(red[0], 1e-5f);
#pragma unroll
  for (int j = 0; j < 4; ++j) {
    float q = rintf(xn[j] * s);
    q = fminf(127.f, fmaxf(-128.f, q));
    xq[(size_t)row * Hc + tid + 256 * j] = (int8_t)q;
    if (writeXn) xnOut[(size_t)row * Hc + tid + 256 * j] = xn[j];
  }
  if (tid == 0) sA[row] = s;
}

__global__ void __launch_bounds__(256)
act_quant_rows(const float* __restrict__ x, int8_t* __restrict__ xq,
               float* __restrict__ sA, int W) {
  const int row = blockIdx.x, tid = threadIdx.x;
  const float* xr = x + (size_t)row * W;
  __shared__ float red[256];
  float am = 0.f;
  for (int i = tid; i < W; i += 256) am = fmaxf(am, fabsf(xr[i]));
  red[tid] = am; __syncthreads();
  for (int o = 128; o > 0; o >>= 1) { if (tid < o) red[tid] = fmaxf(red[tid], red[tid + o]); __syncthreads(); }
  const float s = 127.f / fmaxf(red[0], 1e-5f);
  for (int i = tid; i < W; i += 256) {
    float q = rintf(xr[i] * s);
    q = fminf(127.f, fmaxf(-128.f, q));
    xq[(size_t)row * W + i] = (int8_t)q;
  }
  if (tid == 0) sA[row] = s;
}

static __device__ __forceinline__ void
loadA8(const int8_t* __restrict__ arow, int kb, int h, FragI8& a) {
#pragma unroll
  for (int v = 0; v < 8; ++v)
    a.i[v] = *(const int*)(arow + kb + 32 * (v >> 2) + 16 * h + 4 * (v & 3));
}

static __device__ __forceinline__ void
loadB8(const int8_t* __restrict__ B, int n, int kb, int h, FragI8& b) {
  const int8_t* brow = B + (size_t)n * KD + kb;
#pragma unroll
  for (int v = 0; v < 8; ++v)
    b.i[v] = *(const int*)(brow + 32 * (v >> 2) + 16 * h + 4 * (v & 3));
}

static __device__ __forceinline__ void
bitgemm_k(const int8_t* __restrict__ arow, const int8_t* __restrict__ B,
          int n0, int lane, AccI acc[4]) {
  const int h  = lane >> 4;
  const int nl = lane & 15;
  FragI8 a[2];
  FragI8 b[2][4];

  loadA8(arow, 0, h, a[0]);
#pragma unroll
  for (int t = 0; t < 4; ++t) loadB8(B, n0 + 16 * t + nl, 0, h, b[0][t]);

  for (int kb = 0; kb < KD - 128; kb += 128) {
    loadA8(arow, kb + 64, h, a[1]);
#pragma unroll
    for (int t = 0; t < 4; ++t) loadB8(B, n0 + 16 * t + nl, kb + 64, h, b[1][t]);
#pragma unroll
    for (int t = 0; t < 4; ++t)
      acc[t].v = __builtin_amdgcn_wmma_i32_16x16x64_iu8(
          true, a[0].v, true, b[0][t].v, acc[t].v, false, false);

    loadA8(arow, kb + 128, h, a[0]);
#pragma unroll
    for (int t = 0; t < 4; ++t) loadB8(B, n0 + 16 * t + nl, kb + 128, h, b[0][t]);
#pragma unroll
    for (int t = 0; t < 4; ++t)
      acc[t].v = __builtin_amdgcn_wmma_i32_16x16x64_iu8(
          true, a[1].v, true, b[1][t].v, acc[t].v, false, false);
  }

  loadA8(arow, KD - 64, h, a[1]);
#pragma unroll
  for (int t = 0; t < 4; ++t) loadB8(B, n0 + 16 * t + nl, KD - 64, h, b[1][t]);
#pragma unroll
  for (int t = 0; t < 4; ++t)
    acc[t].v = __builtin_amdgcn_wmma_i32_16x16x64_iu8(
        true, a[0].v, true, b[0][t].v, acc[t].v, false, false);
#pragma unroll
  for (int t = 0; t < 4; ++t)
    acc[t].v = __builtin_amdgcn_wmma_i32_16x16x64_iu8(
        true, a[1].v, true, b[1][t].v, acc[t].v, false, false);
}

__global__ void __launch_bounds__(32)
bitgemm_proj_f16(const int8_t* __restrict__ A, const float* __restrict__ sA,
                 const int8_t* __restrict__ Bw, const float* __restrict__ sW,
                 _Float16* __restrict__ out, int transposedV) {
  const int lane = threadIdx.x, h = lane >> 4, nl = lane & 15;
  const int m0 = blockIdx.x * 16, n0 = blockIdx.y * 64;
  AccI acc[4];
#pragma unroll
  for (int t = 0; t < 4; ++t)
#pragma unroll
    for (int v = 0; v < 8; ++v) acc[t].i[v] = 0;
  bitgemm_k(A + (size_t)(m0 + nl) * KD, Bw, n0, lane, acc);
  const float invW = 1.0f / sW[0];
#pragma unroll
  for (int t = 0; t < 4; ++t)
#pragma unroll
    for (int v = 0; v < 8; ++v) {
      const int row = m0 + v + 8 * h;
      const int col = n0 + 16 * t + nl;
      const float val = (float)acc[t].i[v] * invW / sA[row];
      const int b = row / Sc, s = row % Sc;
      const int head = col >> 6, hd = col & 63;
      const size_t o = transposedV
          ? ((size_t)(b * NHc + head) * HDc + hd) * Sc + s
          : ((size_t)(b * NHc + head) * Sc + s) * HDc + hd;
      out[o] = (_Float16)val;
    }
}

__global__ void __launch_bounds__(32)
bitgemm_oproj(const int8_t* __restrict__ A, const float* __restrict__ sA,
              const int8_t* __restrict__ Bw, const float* __restrict__ sW,
              const float* __restrict__ resid, float* __restrict__ out) {
  const int lane = threadIdx.x, h = lane >> 4, nl = lane & 15;
  const int m0 = blockIdx.x * 16, n0 = blockIdx.y * 64;
  AccI acc[4];
#pragma unroll
  for (int t = 0; t < 4; ++t)
#pragma unroll
    for (int v = 0; v < 8; ++v) acc[t].i[v] = 0;
  bitgemm_k(A + (size_t)(m0 + nl) * KD, Bw, n0, lane, acc);
  const float invW = 1.0f / sW[0];
#pragma unroll
  for (int t = 0; t < 4; ++t)
#pragma unroll
    for (int v = 0; v < 8; ++v) {
      const int row = m0 + v + 8 * h;
      const int col = n0 + 16 * t + nl;
      const size_t o = (size_t)row * Hc + col;
      out[o] = resid[o] + (float)acc[t].i[v] * invW / sA[row];
    }
}

__global__ void __launch_bounds__(32)
flash_attn(const _Float16* __restrict__ Q, const _Float16* __restrict__ Km,
           const _Float16* __restrict__ Vt, float* __restrict__ O) {
  const int lane = threadIdx.x, h = lane >> 4, nl = lane & 15;
  const int q0 = blockIdx.x * 16, head = blockIdx.y, b = blockIdx.z;
  const _Float16* qp = Q  + (size_t)(b * NHc + head) * Sc * HDc;
  const _Float16* kp = Km + (size_t)(b * NHc + head) * Sc * HDc;
  const _Float16* vp = Vt + (size_t)(b * NHc + head) * HDc * Sc;

  FragH aq[2];
#pragma unroll
  for (int f = 0; f < 2; ++f)
#pragma unroll
    for (int v = 0; v < 8; ++v) {
      const int k = 32 * f + 16 * h + 2 * v;
      aq[f].u[v] = *(const unsigned*)(qp + (size_t)(q0 + nl) * HDc + k);
    }

  float mrow[8], lrow[8];
  AccF accO[4];
#pragma unroll
  for (int v = 0; v < 8; ++v) { mrow[v] = -1e30f; lrow[v] = 0.f; }
#pragma unroll
  for (int t = 0; t < 4; ++t)
#pragma unroll
    for (int v = 0; v < 8; ++v) accO[t].f[v] = 0.f;

  __shared__ __align__(16) _Float16 Plds[16][34];
  const float scale = 0.125f;

  for (int kb = 0; kb < q0 + 16; kb += 32) {
    AccF sc[2];
#pragma unroll
    for (int t = 0; t < 2; ++t) {
      FragH bk0, bk1;
      const _Float16* kr = kp + (size_t)(kb + 16 * t + nl) * HDc + 16 * h;
      bk0.q[0] = *(const uint4*)(kr);      bk0.q[1] = *(const uint4*)(kr + 8);
      bk1.q[0] = *(const uint4*)(kr + 32); bk1.q[1] = *(const uint4*)(kr + 40);
      AccF z;
#pragma unroll
      for (int v = 0; v < 8; ++v) z.f[v] = 0.f;
      sc[t].v = __builtin_amdgcn_wmma_f32_16x16x32_f16(
          false, aq[0].v, false, bk0.v, (short)0, z.v, false, false);
      sc[t].v = __builtin_amdgcn_wmma_f32_16x16x32_f16(
          false, aq[1].v, false, bk1.v, (short)0, sc[t].v, false, false);
    }
#pragma unroll
    for (int t = 0; t < 2; ++t)
#pragma unroll
      for (int v = 0; v < 8; ++v) {
        const int key = kb + 16 * t + nl;
        const int qr  = q0 + v + 8 * h;
        const float s = sc[t].f[v] * scale;
        sc[t].f[v] = (key > qr) ? -1e30f : s;
      }
#pragma unroll
    for (int v = 0; v < 8; ++v) {
      float bm = fmaxf(sc[0].f[v], sc[1].f[v]);
#pragma unroll
      for (int o = 8; o > 0; o >>= 1) bm = fmaxf(bm, __shfl_xor(bm, o, 32));
      const float mn = fmaxf(mrow[v], bm);
      const float p0 = __expf(sc[0].f[v] - mn);
      const float p1 = __expf(sc[1].f[v] - mn);
      float rs = p0 + p1;
#pragma unroll
      for (int o = 8; o > 0; o >>= 1) rs += __shfl_xor(rs, o, 32);
      const float alpha = __expf(mrow[v] - mn);
      lrow[v] = lrow[v] * alpha + rs;
      mrow[v] = mn;
#pragma unroll
      for (int t = 0; t < 4; ++t) accO[t].f[v] *= alpha;
      sc[0].f[v] = p0; sc[1].f[v] = p1;
    }
    __syncthreads();
#pragma unroll
    for (int v = 0; v < 8; ++v) {
      Plds[v + 8 * h][nl]      = (_Float16)sc[0].f[v];
      Plds[v + 8 * h][16 + nl] = (_Float16)sc[1].f[v];
    }
    __syncthreads();
    FragH ap;
#pragma unroll
    for (int v = 0; v < 8; ++v) {
      const int k = 16 * h + 2 * v;
      ap.u[v] = *(const unsigned*)(&Plds[nl][k]);
    }
#pragma unroll
    for (int t = 0; t < 4; ++t) {
      FragH bv;
      const _Float16* vr = vp + (size_t)(16 * t + nl) * Sc + kb + 16 * h;
      bv.q[0] = *(const uint4*)(vr); bv.q[1] = *(const uint4*)(vr + 8);
      accO[t].v = __builtin_amdgcn_wmma_f32_16x16x32_f16(
          false, ap.v, false, bv.v, (short)0, accO[t].v, false, false);
    }
  }
#pragma unroll
  for (int t = 0; t < 4; ++t)
#pragma unroll
    for (int v = 0; v < 8; ++v) {
      const int qr  = q0 + v + 8 * h;
      const int col = head * HDc + 16 * t + nl;
      O[((size_t)b * Sc + qr) * Hc + col] = accO[t].f[v] / lrow[v];
    }
}

__global__ void __launch_bounds__(32)
router_argmax(const float* __restrict__ xn, const float* __restrict__ rw,
              int* __restrict__ idx) {
  const int t = blockIdx.x, lane = threadIdx.x;
  const float* xr = xn + (size_t)t * Hc;
  float best = -3.4e38f; int bi = 0;
  for (int e = 0; e < Ec; ++e) {
    const float* wr = rw + (size_t)e * Hc;
    float s = 0.f;
    for (int i = lane; i < Hc; i += 32) s += xr[i] * wr[i];
#pragma unroll
    for (int o = 16; o > 0; o >>= 1) s += __shfl_xor(s, o, 32);
    if (s > best) { best = s; bi = e; }
  }
  if (lane == 0) idx[t] = bi;
}

__global__ void zero_ints(int* p, int n) {
  const int i = blockIdx.x * blockDim.x + threadIdx.x;
  if (i < n) p[i] = 0;
}
__global__ void count_tokens(const int* __restrict__ idx, int* __restrict__ counts) {
  const int t = blockIdx.x * blockDim.x + threadIdx.x;
  if (t < Tc) atomicAdd(&counts[idx[t]], 1);
}
__global__ void scan_offsets(const int* __restrict__ counts,
                             int* __restrict__ offs, int* __restrict__ cursor) {
  if (threadIdx.x == 0) {
    int a = 0;
    for (int e = 0; e < Ec; ++e) { offs[e] = a; cursor[e] = a; a += counts[e]; }
  }
}
__global__ void scatter_tokens(const int* __restrict__ idx,
                               int* __restrict__ cursor, int* __restrict__ list) {
  const int t = blockIdx.x * blockDim.x + threadIdx.x;
  if (t < Tc) { const int p = atomicAdd(&cursor[idx[t]], 1); list[p] = t; }
}

__global__ void __launch_bounds__(32)
bitgemm_gateup(const int8_t* __restrict__ A, const float* __restrict__ sA,
               const int8_t* __restrict__ GW, const int8_t* __restrict__ UW,
               const float* __restrict__ gS, const float* __restrict__ uS,
               const int* __restrict__ counts, const int* __restrict__ offs,
               const int* __restrict__ list, float* __restrict__ hh) {
  const int e = blockIdx.z;
  const int cnt = counts[e];
  const int m0 = blockIdx.x * 16;
  if (m0 >= cnt) return;
  const int lane = threadIdx.x, h = lane >> 4, nl = lane & 15;
  int mi = m0 + nl; if (mi >= cnt) mi = cnt - 1;
  const int tok = list[offs[e] + mi];
  const int8_t* arow = A + (size_t)tok * KD;
  const int8_t* GB = GW + (size_t)e * Dc * Hc;
  const int8_t* UB = UW + (size_t)e * Dc * Hc;
  const int n0 = blockIdx.y * 64;
  AccI ag[4], au[4];
#pragma unroll
  for (int t = 0; t < 4; ++t)
#pragma unroll
    for (int v = 0; v < 8; ++v) { ag[t].i[v] = 0; au[t].i[v] = 0; }
  bitgemm_k(arow, GB, n0, lane, ag);
  bitgemm_k(arow, UB, n0, lane, au);
  const float ig = 1.f / gS[e], iu = 1.f / uS[e];
#pragma unroll
  for (int t = 0; t < 4; ++t)
#pragma unroll
    for (int v = 0; v < 8; ++v) {
      const int r = m0 + v + 8 * h;
      if (r >= cnt) continue;
      const int tok2 = list[offs[e] + r];
      const float isa = 1.f / sA[tok2];
      const float g = (float)ag[t].i[v] * ig * isa;
      const float u = (float)au[t].i[v] * iu * isa;
      const float gr = fmaxf(g, 0.f);
      hh[(size_t)tok2 * Dc + n0 + 16 * t + nl] = gr * gr * u;
    }
}

__global__ void __launch_bounds__(32)
bitgemm_down(const int8_t* __restrict__ A, const float* __restrict__ sA,
             const int8_t* __restrict__ DW, const float* __restrict__ dS,
             const int* __restrict__ counts, const int* __restrict__ offs,
             const int* __restrict__ list, const float* __restrict__ resid,
             float* __restrict__ out) {
  const int e = blockIdx.z;
  const int cnt = counts[e];
  const int m0 = blockIdx.x * 16;
  if (m0 >= cnt) return;
  const int lane = threadIdx.x, h = lane >> 4, nl = lane & 15;
  int mi = m0 + nl; if (mi >= cnt) mi = cnt - 1;
  const int tok = list[offs[e] + mi];
  const int8_t* arow = A + (size_t)tok * KD;
  const int8_t* DB = DW + (size_t)e * Hc * Dc;
  const int n0 = blockIdx.y * 64;
  AccI acc[4];
#pragma unroll
  for (int t = 0; t < 4; ++t)
#pragma unroll
    for (int v = 0; v < 8; ++v) acc[t].i[v] = 0;
  bitgemm_k(arow, DB, n0, lane, acc);
  const float idw = 1.f / dS[e];
#pragma unroll
  for (int t = 0; t < 4; ++t)
#pragma unroll
    for (int v = 0; v < 8; ++v) {
      const int r = m0 + v + 8 * h;
      if (r >= cnt) continue;
      const int tok2 = list[offs[e] + r];
      const int col = n0 + 16 * t + nl;
      const size_t o = (size_t)tok2 * Hc + col;
      out[o] = resid[o] + (float)acc[t].i[v] * idw / sA[tok2];
    }
}

static inline char* carve(char*& p, size_t n) {
  char* r = p;
  p += (n + 255) & ~(size_t)255;
  return r;
}

extern "C" void kernel_launch(void* const* d_in, const int* in_sizes, int n_in,
                              void* d_out, int out_size, void* d_ws, size_t ws_size,
                              hipStream_t stream) {
  (void)in_sizes; (void)n_in; (void)out_size; (void)ws_size;
  const float* x        = (const float*)d_in[0];
  const float* q_w      = (const float*)d_in[1];
  const float* k_w      = (const float*)d_in[2];
  const float* v_w      = (const float*)d_in[3];
  const float* o_w      = (const float*)d_in[4];
  const float* ln1_w    = (const float*)d_in[5];
  const float* ln2_w    = (const float*)d_in[6];
  const float* router_w = (const float*)d_in[7];
  const float* gate_w   = (const float*)d_in[8];
  const float* up_w     = (const float*)d_in[9];
  const float* down_w   = (const float*)d_in[10];
  float* out = (float*)d_out;

  char* p = (char*)d_ws;
  int8_t*    xq1   = (int8_t*)   carve(p, (size_t)Tc * Hc);
  float*     sA1   = (float*)    carve(p, (size_t)Tc * 4);
  int8_t*    w8q   = (int8_t*)   carve(p, (size_t)Hc * Hc);
  int8_t*    w8k   = (int8_t*)   carve(p, (size_t)Hc * Hc);
  int8_t*    w8v   = (int8_t*)   carve(p, (size_t)Hc * Hc);
  int8_t*    w8o   = (int8_t*)   carve(p, (size_t)Hc * Hc);
  float*     wS    = (float*)    carve(p, 4 * 4);
  int8_t*    gw8   = (int8_t*)   carve(p, (size_t)Ec * Dc * Hc);
  int8_t*    uw8   = (int8_t*)   carve(p, (size_t)Ec * Dc * Hc);
  int8_t*    dw8   = (int8_t*)   carve(p, (size_t)Ec * Hc * Dc);
  float*     eS    = (float*)    carve(p, 3 * Ec * 4);
  float*     part  = (float*)    carve(p, (size_t)Ec * PB * 4);
  _Float16*  qb    = (_Float16*) carve(p, (size_t)Tc * Hc * 2);
  _Float16*  kbuf  = (_Float16*) carve(p, (size_t)Tc * Hc * 2);
  _Float16*  vtb   = (_Float16*) carve(p, (size_t)Tc * Hc * 2);
  float*     hbuf  = (float*)    carve(p, (size_t)Tc * Hc * 4);
  int8_t*    hq1   = (int8_t*)   carve(p, (size_t)Tc * Hc);
  float*     sH1   = (float*)    carve(p, (size_t)Tc * 4);
  float*     x1    = (float*)    carve(p, (size_t)Tc * Hc * 4);
  float*     xn2   = (float*)    carve(p, (size_t)Tc * Hc * 4);
  int8_t*    xq2   = (int8_t*)   carve(p, (size_t)Tc * Hc);
  float*     sA2   = (float*)    carve(p, (size_t)Tc * 4);
  int*       idx   = (int*)      carve(p, (size_t)Tc * 4);
  int*       cnts  = (int*)      carve(p, Ec * 4);
  int*       offs  = (int*)      carve(p, Ec * 4);
  int*       curs  = (int*)      carve(p, Ec * 4);
  int*       list  = (int*)      carve(p, (size_t)Tc * 4);
  float*     hh    = (float*)    carve(p, (size_t)Tc * Dc * 4);
  int8_t*    hq2   = (int8_t*)   carve(p, (size_t)Tc * Dc);
  float*     sH2   = (float*)    carve(p, (size_t)Tc * 4);

  const float* attw[4] = {q_w, k_w, v_w, o_w};
  int8_t* att8[4] = {w8q, w8k, w8v, w8o};
  const long HH = (long)Hc * Hc;
  for (int i = 0; i < 4; ++i) {
    absmean_partial<<<dim3(PB, 1), 256, 0, stream>>>(attw[i], HH, part);
    make_scales<<<1, 256, 0, stream>>>(part, 1.0f / (float)HH, wS + i);
    quantize_w<<<1024, 256, 0, stream>>>(attw[i], wS + i, HH, HH, att8[i]);
  }
  const long EH = (long)Dc * Hc;
  const long ET = (long)Ec * EH;
  const float* expw[3] = {gate_w, up_w, down_w};
  int8_t* exp8[3] = {gw8, uw8, dw8};
  for (int i = 0; i < 3; ++i) {
    absmean_partial<<<dim3(PB, Ec), 256, 0, stream>>>(expw[i], EH, part);
    make_scales<<<Ec, 256, 0, stream>>>(part, 1.0f / (float)EH, eS + i * Ec);
    quantize_w<<<2048, 256, 0, stream>>>(expw[i], eS + i * Ec, EH, ET, exp8[i]);
  }

  rmsnorm_quant<<<Tc, 256, 0, stream>>>(x, ln1_w, xn2, 0, xq1, sA1);
  dim3 gg(Tc / 16, Hc / 64);
  bitgemm_proj_f16<<<gg, 32, 0, stream>>>(xq1, sA1, w8q, wS + 0, qb,  0);
  bitgemm_proj_f16<<<gg, 32, 0, stream>>>(xq1, sA1, w8k, wS + 1, kbuf, 0);
  bitgemm_proj_f16<<<gg, 32, 0, stream>>>(xq1, sA1, w8v, wS + 2, vtb, 1);
  flash_attn<<<dim3(Sc / 16, NHc, Bc), 32, 0, stream>>>(qb, kbuf, vtb, hbuf);
  act_quant_rows<<<Tc, 256, 0, stream>>>(hbuf, hq1, sH1, Hc);
  bitgemm_oproj<<<gg, 32, 0, stream>>>(hq1, sH1, w8o, wS + 3, x, x1);

  rmsnorm_quant<<<Tc, 256, 0, stream>>>(x1, ln2_w, xn2, 1, xq2, sA2);
  router_argmax<<<Tc, 32, 0, stream>>>(xn2, router_w, idx);
  zero_ints<<<1, 32, 0, stream>>>(cnts, Ec);
  count_tokens<<<Tc / 256, 256, 0, stream>>>(idx, cnts);
  scan_offsets<<<1, 1, 0, stream>>>(cnts, offs, curs);
  scatter_tokens<<<Tc / 256, 256, 0, stream>>>(idx, curs, list);
  bitgemm_gateup<<<dim3(Tc / 16, Dc / 64, Ec), 32, 0, stream>>>(
      xq2, sA2, gw8, uw8, eS, eS + Ec, cnts, offs, list, hh);
  act_quant_rows<<<Tc, 256, 0, stream>>>(hh, hq2, sH2, Dc);
  bitgemm_down<<<dim3(Tc / 16, Hc / 64, Ec), 32, 0, stream>>>(
      hq2, sH2, dw8, eS + 2 * Ec, cnts, offs, list, x1, out);
}
